// RNN_84026740179641
// MI455X (gfx1250) — matrix operand order repaired and checked by reading, not compiled, not run on hardware
//
#include <hip/hip_runtime.h>
#include <math.h>

// Problem dims
#define T_DIM  64
#define S_DIM  64
#define D_DIM  2048
#define H_DIM  2048
#define NSTEP  (T_DIM * S_DIM)      // 4096 serial RNN steps
#define CAT    (D_DIM + H_DIM)      // 4096 (W1 row length)
#define OUT_DIM 2
#define RNN_BLOCKS  32
#define RNN_THREADS 512

typedef __attribute__((ext_vector_type(16))) __bf16 v16bf;
typedef __attribute__((ext_vector_type(8)))  __bf16 v8bf;
typedef __attribute__((ext_vector_type(8)))  float  v8f;

union BFrag { v16bf full; v8bf part[2]; };

// ---------------------------------------------------------------------------
// Small prep: X fp32 -> bf16, h0 init, barrier counter reset.
// ---------------------------------------------------------------------------
__global__ void __launch_bounds__(256)
conv_kernel(const float* __restrict__ X,
            const float* __restrict__ hidden,
            __bf16* __restrict__ Xb,
            float* __restrict__ h0,
            unsigned* __restrict__ ctr) {
    const long tid  = (long)blockIdx.x * blockDim.x + threadIdx.x;
    const long nthr = (long)gridDim.x * blockDim.x;
    const long NX = (long)NSTEP * D_DIM;          // 8,388,608
    for (long i = tid; i < NX; i += nthr) Xb[i] = (__bf16)X[i];
    for (long i = tid; i < H_DIM; i += nthr) h0[i] = hidden[i];
    if (tid == 0) *ctr = 0u;
}

// ---------------------------------------------------------------------------
// LDS-tiled transpose + fp32->bf16: outT[k][n] = W1[n][colOff + k].
// 64x64 tiles; LDS padded [64][66] so transposed reads are bank-conflict-free
// (row stride 132 B = 33 dwords, gcd(33,64)=1).
// ---------------------------------------------------------------------------
__global__ void __launch_bounds__(256)
transpose_kernel(const float* __restrict__ W1,
                 __bf16* __restrict__ outT,
                 int colOff) {
    __shared__ __bf16 tile[64][66];
    const int bx = blockIdx.x & 31;          // k-tile index
    const int by = blockIdx.x >> 5;          // n-tile index
    const int k0 = bx * 64, n0 = by * 64;
    const int tx = threadIdx.x & 63;
    const int ty = threadIdx.x >> 6;         // 0..3
#pragma unroll
    for (int i = 0; i < 64; i += 4) {        // coalesced read of W1 rows
        const int n = n0 + ty + i;
        tile[ty + i][tx] = (__bf16)W1[(long)n * CAT + colOff + k0 + tx];
    }
    __syncthreads();
#pragma unroll
    for (int i = 0; i < 64; i += 4) {        // coalesced write of outT rows
        const int k = k0 + ty + i;
        outT[(long)k * H_DIM + n0 + tx] = tile[tx][ty + i];
    }
}

// ---------------------------------------------------------------------------
// WMMA GEMM: P[4096,2048] = Xb @ WxT + b1   (bf16 in, f32 accumulate)
// One wave computes a 32x64 strip: 2 A-frags x 4 B-frags -> 8 WMMAs / k-step.
// ---------------------------------------------------------------------------
__global__ void __launch_bounds__(256)
gemm_xw_kernel(const __bf16* __restrict__ A,    // [4096, 2048] row-major (M x K)
               const __bf16* __restrict__ B,    // [2048, 2048] row-major (K x N)
               const float*  __restrict__ bias, // [2048]
               float*        __restrict__ C) {  // [4096, 2048]
    const int lane   = threadIdx.x & 31;
    const int wave   = threadIdx.x >> 5;
    const int waveId = blockIdx.x * 8 + wave;
    const int mt = waveId >> 5;                  // 0..127 (32-row tiles)
    const int ng = waveId & 31;                  // 0..31  (64-col groups)
    const int m0 = mt * 32;
    const int n0 = ng * 64;

    // A fragment addressing (16-bit A 16x32 ISA layout):
    //   lane<16: row M=lane,     K chunks {0..7} and {16..23}
    //   lane>=16: row M=lane-16, K chunks {8..15} and {24..31}
    const int aRow0 = m0 + (lane & 15);
    const int aRow1 = aRow0 + 16;
    const int aKoff = (lane < 16) ? 0 : 8;

    v8f acc[2][4] = {};

    for (int k0 = 0; k0 < D_DIM; k0 += 32) {
        BFrag a0, a1;
        const __bf16* ap0 = A + (long)aRow0 * D_DIM + (k0 + aKoff);
        const __bf16* ap1 = A + (long)aRow1 * D_DIM + (k0 + aKoff);
        a0.part[0] = *(const v8bf*)(ap0);  a0.part[1] = *(const v8bf*)(ap0 + 16);
        a1.part[0] = *(const v8bf*)(ap1);  a1.part[1] = *(const v8bf*)(ap1 + 16);

        // B fragment: N-col = lane&15, K in same order as A (aKoff, +16).
        const __bf16* bp = B + (long)(k0 + aKoff) * H_DIM + n0 + (lane & 15);
        BFrag b[4];
#pragma unroll
        for (int j = 0; j < 4; ++j) {
#pragma unroll
            for (int s = 0; s < 8; ++s) {
                b[j].full[s]     = bp[(long)s * H_DIM + j * 16];
                b[j].full[s + 8] = bp[(long)(s + 16) * H_DIM + j * 16];
            }
        }
#pragma unroll
        for (int j = 0; j < 4; ++j) {
            acc[0][j] = __builtin_amdgcn_wmma_f32_16x16x32_bf16(
                false, a0.full, false, b[j].full, (short)0, acc[0][j], false, false);
            acc[1][j] = __builtin_amdgcn_wmma_f32_16x16x32_bf16(
                false, a1.full, false, b[j].full, (short)0, acc[1][j], false, false);
        }
    }

    // C layout: VGPR r -> row (tile_m + r), +8 for lanes 16..31; col = lane&15
    const int col = lane & 15;
#pragma unroll
    for (int i = 0; i < 2; ++i) {
        const int rowBase = m0 + i * 16 + ((lane < 16) ? 0 : 8);
#pragma unroll
        for (int j = 0; j < 4; ++j) {
            const int n  = n0 + j * 16 + col;
            const float bv = bias[n];
#pragma unroll
            for (int r = 0; r < 8; ++r) {
                C[(long)(rowBase + r) * H_DIM + n] = acc[i][j][r] + bv;
            }
        }
    }
}

// ---------------------------------------------------------------------------
// Serial recurrence via WMMA with LDS-resident Wh slice.
// 32 persistent blocks x 512 threads. Block owns 64 N-columns of Wh:
// 2048x64 bf16 = 256 KB staged ONCE in LDS, pre-swizzled into B-fragment
// order [k-step][n-tile][lane][32B] so every ds_load_b128 is conflict-free.
// Per step: A-frag = h broadcast (bf16 in LDS), 16 waves cover 4 n-tiles x
// 4 k-chunks; LDS reduce; tanh; global arrive/spin barrier.
// ---------------------------------------------------------------------------
__global__ void __launch_bounds__(RNN_THREADS)
rnn_kernel(const float*  __restrict__ P,     // [4096, 2048], b1 folded in
           const __bf16* __restrict__ WhT,   // [2048, 2048]  (K x N)
           float* h0, float* h1,             // ping-pong hidden buffers [2048]
           float* __restrict__ U,            // [64, 2048] tweet vectors
           unsigned* ctr) {
    __shared__ __bf16 sWh[(H_DIM / 32) * 4 * 32 * 16];  // 256 KB fragment store
    __shared__ __bf16 shb[H_DIM];                       // 4 KB staged hidden
    __shared__ float  spart[16][16];
    volatile unsigned* vctr = (volatile unsigned*)ctr;

    const int tid  = threadIdx.x;
    const int lane = tid & 31;
    const int w    = tid >> 5;               // 0..15
    const int nt   = w >> 2;                 // n-tile within block (0..3)
    const int kc   = w & 3;                  // k-chunk (0..3), 16 k-steps each
    const int n0base = blockIdx.x * 64;
    const int aoff = (lane < 16) ? 0 : 8;

    // ---- one-time fill: swizzle this block's Wh slice into fragment order
    // global element (k, nl) -> frag slot ((ks*4 + nt)*32 + (kr/8%2)*16 + c)*16 + kr%8 + 8*(kr/16)
    //   k = ks*32 + kr, nl = nt*16 + c ; coalesced global reads over nl.
    for (int idx = tid; idx < H_DIM * 64; idx += RNN_THREADS) {
        const int k  = idx >> 6;
        const int nl = idx & 63;
        const int ks = k >> 5, kr = k & 31;
        const int tnt = nl >> 4, c = nl & 15;
        sWh[((ks * 4 + tnt) * 32 + ((kr >> 3) & 1) * 16 + c) * 16 + (kr & 7) + ((kr >> 4) << 3)] =
            WhT[(long)k * H_DIM + n0base + nl];
    }
    __syncthreads();

    for (int t = 0; t < NSTEP; ++t) {
        const float* hc = (t & 1) ? h1 : h0;
        float*       hn = (t & 1) ? h0 : h1;

        // stage current hidden into LDS as bf16
        for (int i = tid; i < H_DIM; i += RNN_THREADS) shb[i] = (__bf16)hc[i];
        __syncthreads();

        v8f acc = {};
        for (int ks = kc * 16; ks < kc * 16 + 16; ++ks) {
            const int k0 = ks * 32;
            BFrag a;                          // identical rows = broadcast h
            a.part[0] = *(const v8bf*)(&shb[k0 + aoff]);
            a.part[1] = *(const v8bf*)(&shb[k0 + aoff + 16]);
            BFrag b;                          // conflict-free fragment fetch
            const __bf16* bp = &sWh[((ks * 4 + nt) * 32 + lane) * 16];
            b.part[0] = *(const v8bf*)(bp);
            b.part[1] = *(const v8bf*)(bp + 8);
            acc = __builtin_amdgcn_wmma_f32_16x16x32_bf16(
                false, a.full, false, b.full, (short)0, acc, false, false);
        }
        if (lane < 16) spart[w][lane] = acc[0];   // row M=0 of D
        __syncthreads();

        if (tid < 64) {
            const int tnt = tid >> 4;             // n-tile 0..3
            const int col = tid & 15;
            const int n = n0base + tnt * 16 + col;
            float s = P[(long)t * H_DIM + n];
#pragma unroll
            for (int c = 0; c < 4; ++c) s += spart[tnt * 4 + c][col];
            const float hv = tanhf(s);
            hn[n] = hv;
            if ((t & (S_DIM - 1)) == (S_DIM - 1))
                U[(long)(t >> 6) * H_DIM + n] = hv;   // tweet vector
        }

        // grid-wide barrier: publish hn, arrive, spin to generation t+1
        __threadfence();
        __syncthreads();
        if (tid == 0) {
            atomicAdd(ctr, 1u);
            const unsigned target = (unsigned)(t + 1) * RNN_BLOCKS;
            while (*vctr < target) { __builtin_amdgcn_s_sleep(1); }
        }
        __syncthreads();
        __threadfence();
    }
}

// ---------------------------------------------------------------------------
// Output head: out[o] = dot(U.flat, W2[o]) + b2[o], o = 0,1
// ---------------------------------------------------------------------------
__global__ void __launch_bounds__(256)
out_kernel(const float* __restrict__ U,      // [131072]
           const float* __restrict__ W2,     // [2, 131072]
           const float* __restrict__ b2,
           float* __restrict__ out) {
    __shared__ float red[256];
    const int  o = blockIdx.x;
    const long N = (long)T_DIM * H_DIM;
    float s = 0.f;
    for (long i = threadIdx.x; i < N; i += 256)
        s += U[i] * W2[(long)o * N + i];
    red[threadIdx.x] = s;
    __syncthreads();
    for (int w = 128; w > 0; w >>= 1) {
        if (threadIdx.x < w) red[threadIdx.x] += red[threadIdx.x + w];
        __syncthreads();
    }
    if (threadIdx.x == 0) out[o] = red[0] + b2[o];
}

// ---------------------------------------------------------------------------
extern "C" void kernel_launch(void* const* d_in, const int* in_sizes, int n_in,
                              void* d_out, int out_size, void* d_ws, size_t ws_size,
                              hipStream_t stream) {
    const float* X   = (const float*)d_in[0];   // [64,64,2048]
    const float* hid = (const float*)d_in[1];   // [2048]
    const float* W1  = (const float*)d_in[2];   // [2048,4096]
    const float* b1  = (const float*)d_in[3];   // [2048]
    const float* W2  = (const float*)d_in[4];   // [2,131072]
    const float* b2  = (const float*)d_in[5];   // [2]
    float* out = (float*)d_out;                 // [2]

    char* ws = (char*)d_ws;
    __bf16*   Xb  = (__bf16*)(ws + 0);                 // 16 MB
    __bf16*   WxT = (__bf16*)(ws + 16777216);          //  8 MB
    __bf16*   WhT = (__bf16*)(ws + 25165824);          //  8 MB
    float*    P   = (float*) (ws + 33554432);          // 32 MB
    float*    h0  = (float*) (ws + 67108864);          //  8 KB
    float*    h1  = (float*) (ws + 67117056);          //  8 KB
    float*    U   = (float*) (ws + 67125248);          // 512 KB
    unsigned* ctr = (unsigned*)(ws + 67649536);

    conv_kernel<<<512, 256, 0, stream>>>(X, hid, Xb, h0, ctr);
    transpose_kernel<<<1024, 256, 0, stream>>>(W1, WxT, 0);       // Wx half
    transpose_kernel<<<1024, 256, 0, stream>>>(W1, WhT, D_DIM);   // Wh half
    gemm_xw_kernel<<<512, 256, 0, stream>>>(Xb, WxT, b1, P);      // 4096 waves = 128x32 tiles
    rnn_kernel<<<RNN_BLOCKS, RNN_THREADS, 0, stream>>>(P, WhT, h0, h1, U, ctr);
    out_kernel<<<OUT_DIM, 256, 0, stream>>>(U, W2, b2, out);
}
